// SelfAttention1D_48979807043882
// MI455X (gfx1250) — compile-verified, matrix operand order checked by reading, not run on hardware
//
#include <hip/hip_runtime.h>
#include <hip/hip_bf16.h>
#include <math.h>

// ---------------------------------------------------------------------------
// SelfAttention1D for MI455X (gfx1250, wave32, WMMA bf16 16x16x32 + TDM)
// b=8, c=256, l=2048, heads=4, head_dim=64, groups=8
// ---------------------------------------------------------------------------

typedef __bf16 bf16;
typedef __attribute__((ext_vector_type(16))) __bf16 bf16x16;
typedef __attribute__((ext_vector_type(8)))  float  f32x8;
typedef unsigned int u32;
typedef __attribute__((ext_vector_type(4))) u32 u32x4;
typedef __attribute__((ext_vector_type(4))) int i32x4;
typedef __attribute__((ext_vector_type(8))) int i32x8;

#define B_  8
#define C_  256
#define L_  2048
#define H_  4
#define D_  64
#define G_  8
#define GC_ 32
#define GN_ELEMS (GC_ * L_)

// ---- CDNA5 feature detection ----------------------------------------------
#if defined(__has_builtin)
#  if __has_builtin(__builtin_amdgcn_tensor_load_to_lds)
#    define USE_TDM 1
#  endif
#endif
#ifndef USE_TDM
#  define USE_TDM 0
#endif

#if defined(__has_builtin) && __has_builtin(__builtin_amdgcn_s_wait_tensorcnt)
#  define WAIT_TENSORCNT0() __builtin_amdgcn_s_wait_tensorcnt(0)
#else
#  define WAIT_TENSORCNT0() asm volatile("s_wait_tensorcnt 0x0" ::: "memory")
#endif

__device__ __forceinline__ f32x8 wmma_bf16(bf16x16 a, bf16x16 b, f32x8 c) {
    return __builtin_amdgcn_wmma_f32_16x16x32_bf16(
        false, a, false, b, (short)0, c, false, false);
}

#if USE_TDM
// Issue one 2D TDM tile load (bf16 elements) global -> LDS.
// D# per CDNA5 ISA: group0 = {flags, lds_addr,
// global_addr, type=2}; group1 = {data_size=1(2B), tensor/tile dims, stride}.
__device__ __forceinline__ void tdm_load_2d_bf16(u32 lds_off, const void* gaddr,
                                                 u32 tile_d0, u32 tile_d1,
                                                 u32 row_stride_elems) {
    unsigned long long ga = (unsigned long long)(size_t)gaddr;
    u32x4 g0;
    g0.x = 0x1u;                                        // count=1, user descriptor
    g0.y = lds_off;                                     // lds_addr (bytes)
    g0.z = (u32)(ga & 0xffffffffu);                     // global_addr[31:0]
    g0.w = (u32)((ga >> 32) & 0x1ffffffu) | (2u << 30); // addr[56:32] | type=2
    i32x8 g1;
    g1[0] = (int)(1u << 16);                            // wg_mask=0, data_size=2B
    g1[1] = (int)((tile_d0 & 0xffffu) << 16);           // tensor_dim0 = tile_d0
    g1[2] = (int)((tile_d1 & 0xffffu) << 16);           // tensor_dim1 = tile_d1
    g1[3] = (int)((tile_d0 & 0xffffu) << 16);           // tile_dim0
    g1[4] = (int)(tile_d1 & 0xffffu);                   // tile_dim1 (tile_dim2=0)
    g1[5] = (int)row_stride_elems;                      // tensor_dim0_stride lo
    g1[6] = 0;                                          // stride hi / dim1_stride
    g1[7] = 0;
    i32x4 z4 = {0, 0, 0, 0};
#if __clang_major__ >= 23
    i32x8 z8 = {0, 0, 0, 0, 0, 0, 0, 0};
    __builtin_amdgcn_tensor_load_to_lds(g0, g1, z4, z4, z8, 0);
#else
    __builtin_amdgcn_tensor_load_to_lds(g0, g1, z4, z4, 0);
#endif
}
#endif // USE_TDM

// ---------------------------------------------------------------------------
// 0) fp32 -> bf16 conversion
// ---------------------------------------------------------------------------
__global__ void cvt_bf16_kernel(const float* __restrict__ in,
                                bf16* __restrict__ out, int n) {
    int i = blockIdx.x * 256 + threadIdx.x;
    if (i < n) out[i] = (bf16)in[i];
}

// ---------------------------------------------------------------------------
// 1) GroupNorm -> bf16, stored transposed: xnT[b][l][c]
// ---------------------------------------------------------------------------
__global__ void gn_kernel(const float* __restrict__ x,
                          const float* __restrict__ gamma,
                          const float* __restrict__ beta,
                          bf16* __restrict__ xnT) {
    const int b   = blockIdx.x >> 3;
    const int g   = blockIdx.x & 7;
    const int tid = threadIdx.x;
    const float* xg = x + ((size_t)b * C_ + g * GC_) * L_;

    float s = 0.f, ss = 0.f;
    for (int i = tid; i < GN_ELEMS; i += 256) {
        float v = xg[i];
        s += v; ss += v * v;
    }
    __shared__ float rs[256], rss[256];
    rs[tid] = s; rss[tid] = ss;
    __syncthreads();
    for (int off = 128; off > 0; off >>= 1) {
        if (tid < off) { rs[tid] += rs[tid + off]; rss[tid] += rss[tid + off]; }
        __syncthreads();
    }
    const float inv_n = 1.f / (float)GN_ELEMS;
    const float mean  = rs[0] * inv_n;
    const float var   = rss[0] * inv_n - mean * mean;
    const float rstd  = rsqrtf(var + 1e-5f);

    __shared__ __align__(16) float tile[GC_][65];
    const int nr = tid & 63, cr = tid >> 6;
    const int cw = tid & 31, nw0 = tid >> 5;
    for (int t = 0; t < L_ / 64; ++t) {
        const int n0 = t * 64;
        __syncthreads();
        #pragma unroll
        for (int cs = 0; cs < 8; ++cs) {
            const int ch = cs * 4 + cr;
            const int gc = g * GC_ + ch;
            float v = xg[(size_t)ch * L_ + n0 + nr];
            tile[ch][nr] = (v - mean) * rstd * gamma[gc] + beta[gc];
        }
        __syncthreads();
        #pragma unroll
        for (int ns = 0; ns < 8; ++ns) {
            const int nn = ns * 8 + nw0;
            xnT[((size_t)b * L_ + n0 + nn) * C_ + g * GC_ + cw] = (bf16)tile[cw][nn];
        }
    }
}

// ---------------------------------------------------------------------------
// 2) QKV GEMM (M=768, N=2048, K=256); wave tile 16x64.
//    Outputs scattered to attention layouts: qT/kT [b,h,l,d], v [b,h,d,l].
// ---------------------------------------------------------------------------
__global__ void qkv_gemm_kernel(const bf16* __restrict__ W,
                                const bf16* __restrict__ xnT,
                                bf16* __restrict__ qT,
                                bf16* __restrict__ kT,
                                bf16* __restrict__ vv) {
    const int wid  = blockIdx.x * 8 + (threadIdx.x >> 5);
    const int lane = threadIdx.x & 31;
    const int hf   = lane >> 4;
    const int lm   = lane & 15;
    const int batch = wid / 1536;
    const int rem   = wid % 1536;
    const int m0    = (rem >> 5) * 16;
    const int n0    = (rem & 31) * 64;
    const bf16* xb = xnT + (size_t)batch * L_ * C_;

    f32x8 acc[4] = {};
    #pragma unroll 2
    for (int ks = 0; ks < C_ / 32; ++ks) {
        const int k0 = ks * 32 + hf * 16;
        // batch all fragment loads first -> one load clause, staged waits
        bf16x16 a = *reinterpret_cast<const bf16x16*>(W + (size_t)(m0 + lm) * C_ + k0);
        bf16x16 bfr[4];
        #pragma unroll
        for (int j = 0; j < 4; ++j)
            bfr[j] = *reinterpret_cast<const bf16x16*>(
                xb + (size_t)(n0 + j * 16 + lm) * C_ + k0);
        if (ks + 1 < C_ / 32)
            __builtin_prefetch(xb + (size_t)(n0 + lm) * C_ + k0 + 32, 0, 1);
        #pragma unroll
        for (int j = 0; j < 4; ++j)
            acc[j] = wmma_bf16(a, bfr[j], acc[j]);
    }

    #pragma unroll
    for (int j = 0; j < 4; ++j) {
        #pragma unroll
        for (int r = 0; r < 8; ++r) {
            const int m = m0 + r + 8 * hf;
            const int n = n0 + j * 16 + lm;
            const bf16 val = (bf16)acc[j][r];
            if (m < 256) {
                const int head = m >> 6, dd = m & 63;
                qT[((size_t)(batch * H_ + head) * L_ + n) * D_ + dd] = val;
            } else if (m < 512) {
                const int mm = m - 256, head = mm >> 6, dd = mm & 63;
                kT[((size_t)(batch * H_ + head) * L_ + n) * D_ + dd] = val;
            } else {
                const int mm = m - 512, head = mm >> 6, dd = mm & 63;
                vv[((size_t)(batch * H_ + head) * D_ + dd) * L_ + n] = val;
            }
        }
    }
}

// ---------------------------------------------------------------------------
// 3) Flash attention, 4 waves x 16 queries per block.
//    K/V 64x64 tiles staged in LDS via TDM (double-buffered, TENSORcnt),
//    shared by all 4 waves; S = QK^T/8 with online softmax; O += P V.
// ---------------------------------------------------------------------------
__global__ void attn_kernel(const bf16* __restrict__ qT,
                            const bf16* __restrict__ kT,
                            const bf16* __restrict__ vv,
                            bf16* __restrict__ attnT) {
    const int bh_blocks = L_ / 64;
    const int b_   = blockIdx.x / (H_ * bh_blocks);
    const int rem  = blockIdx.x % (H_ * bh_blocks);
    const int head = rem / bh_blocks;
    const int it   = rem % bh_blocks;
    const int wave = threadIdx.x >> 5;
    const int lane = threadIdx.x & 31;
    const int hf   = lane >> 4;
    const int lm   = lane & 15;
    const int i0   = it * 64 + wave * 16;

    const bf16* qb = qT + (size_t)(b_ * H_ + head) * L_ * D_;
    const bf16* kb = kT + (size_t)(b_ * H_ + head) * L_ * D_;
    const bf16* vb = vv + (size_t)(b_ * H_ + head) * D_ * L_;
    const float scale = 0.125f;

    // LDS: double-buffered K/V tiles (shared by the block) + per-wave P slice
    __shared__ __align__(32) bf16 kvbuf[2][2][64][64];   // 32 KB
    __shared__ __align__(32) bf16 pbuf[4][16][64];       //  8 KB
    bf16 (*myp)[64] = pbuf[wave];

    // Q fragments held in VGPRs for the whole kernel
    bf16x16 aq[2];
    #pragma unroll
    for (int ks = 0; ks < 2; ++ks)
        aq[ks] = *reinterpret_cast<const bf16x16*>(
            qb + (size_t)(i0 + lm) * D_ + ks * 32 + hf * 16);

    f32x8 o_acc[4] = {};
    float mrow[8], lrow[8];
    #pragma unroll
    for (int r = 0; r < 8; ++r) { mrow[r] = -1e30f; lrow[r] = 0.f; }

    // ---- tile fill: TDM (wave 0 issues DMA) or cooperative loads ----
    auto fill_tiles = [&](int buf, int j0n) {
#if USE_TDM
        if (threadIdx.x < 32) {
            // K tile: 64 token-rows of kT (row pitch D_), 64 bf16 each
            tdm_load_2d_bf16((u32)(size_t)(void*)&kvbuf[buf][0][0][0],
                             kb + (size_t)j0n * D_, 64, 64, D_);
            // V tile: 64 d-rows of v (row pitch L_), cols j0n..j0n+63
            tdm_load_2d_bf16((u32)(size_t)(void*)&kvbuf[buf][1][0][0],
                             vb + j0n, 64, 64, L_);
        }
#else
        // cooperative fill: 128 threads x 4 x 16B each per tile
        for (int t = threadIdx.x; t < 64 * 8; t += 128) {
            const int row = t >> 3, c8 = (t & 7) * 8;
            *(uint4*)&kvbuf[buf][0][row][c8] =
                *(const uint4*)(kb + (size_t)(j0n + row) * D_ + c8);
            *(uint4*)&kvbuf[buf][1][row][c8] =
                *(const uint4*)(vb + (size_t)row * L_ + j0n + c8);
        }
#endif
    };

    fill_tiles(0, 0);
    int cur = 0;

    for (int jb = 0; jb < L_ / 64; ++jb) {
#if USE_TDM
        if (threadIdx.x < 32) WAIT_TENSORCNT0();
#endif
        __syncthreads();                         // kvbuf[cur] ready for all waves
        if (jb + 1 < L_ / 64) fill_tiles(cur ^ 1, (jb + 1) * 64);  // overlap next DMA

        bf16 (*ktile)[64] = kvbuf[cur][0];
        bf16 (*vtile)[64] = kvbuf[cur][1];

        // ---- S = Q^T K (16 x 64) ----
        f32x8 s[4] = {};
        #pragma unroll
        for (int ks = 0; ks < 2; ++ks) {
            bf16x16 bk[4];
            #pragma unroll
            for (int jt = 0; jt < 4; ++jt)
                bk[jt] = *reinterpret_cast<const bf16x16*>(
                    &ktile[jt * 16 + lm][ks * 32 + hf * 16]);
            #pragma unroll
            for (int jt = 0; jt < 4; ++jt)
                s[jt] = wmma_bf16(aq[ks], bk[jt], s[jt]);
        }

        // ---- online softmax (row = r + 8*hf) ----
        #pragma unroll
        for (int r = 0; r < 8; ++r) {
            float mx = fmaxf(fmaxf(s[0][r], s[1][r]), fmaxf(s[2][r], s[3][r]));
            #pragma unroll
            for (int msk = 1; msk <= 8; msk <<= 1)
                mx = fmaxf(mx, __shfl_xor(mx, msk, 32));
            const float mn   = fmaxf(mrow[r], mx * scale);
            const float corr = __expf(mrow[r] - mn);
            mrow[r] = mn;
            lrow[r] *= corr;
            o_acc[0][r] *= corr; o_acc[1][r] *= corr;
            o_acc[2][r] *= corr; o_acc[3][r] *= corr;

            float psum = 0.f;
            #pragma unroll
            for (int jt = 0; jt < 4; ++jt) {
                float p = __expf(s[jt][r] * scale - mn);
                psum += p;
                myp[r + 8 * hf][jt * 16 + lm] = (bf16)p;
            }
            #pragma unroll
            for (int msk = 1; msk <= 8; msk <<= 1)
                psum += __shfl_xor(psum, msk, 32);
            lrow[r] += psum;
        }

        // wave-private P slice: C-layout stores -> A-layout reads
        asm volatile("s_wait_dscnt 0x0" ::: "memory");

        // ---- O += P V ----
        #pragma unroll
        for (int ks = 0; ks < 2; ++ks) {
            bf16x16 ap = *reinterpret_cast<const bf16x16*>(
                &myp[lm][ks * 32 + hf * 16]);
            bf16x16 bv[4];
            #pragma unroll
            for (int dt = 0; dt < 4; ++dt)
                bv[dt] = *reinterpret_cast<const bf16x16*>(
                    &vtile[dt * 16 + lm][ks * 32 + hf * 16]);
            #pragma unroll
            for (int dt = 0; dt < 4; ++dt)
                o_acc[dt] = wmma_bf16(ap, bv[dt], o_acc[dt]);
        }

        __syncthreads();                         // all waves done with kvbuf[cur]
        cur ^= 1;
    }

    #pragma unroll
    for (int dt = 0; dt < 4; ++dt) {
        #pragma unroll
        for (int r = 0; r < 8; ++r) {
            const int i  = i0 + r + 8 * hf;
            const int cc = head * D_ + dt * 16 + lm;
            attnT[((size_t)b_ * L_ + i) * C_ + cc] = (bf16)(o_acc[dt][r] / lrow[r]);
        }
    }
}

// ---------------------------------------------------------------------------
// 4) Output projection + bias + residual (fp32 out)
// ---------------------------------------------------------------------------
__global__ void proj_gemm_kernel(const bf16* __restrict__ Pw,
                                 const bf16* __restrict__ attnT,
                                 const float* __restrict__ bias,
                                 const float* __restrict__ xres,
                                 float* __restrict__ out) {
    const int wid  = blockIdx.x * 8 + (threadIdx.x >> 5);
    const int lane = threadIdx.x & 31;
    const int hf   = lane >> 4;
    const int lm   = lane & 15;
    const int batch = wid / 512;
    const int rem   = wid % 512;
    const int m0    = (rem >> 5) * 16;
    const int n0    = (rem & 31) * 64;
    const bf16* ab = attnT + (size_t)batch * L_ * C_;

    f32x8 acc[4] = {};
    #pragma unroll 2
    for (int ks = 0; ks < C_ / 32; ++ks) {
        const int k0 = ks * 32 + hf * 16;
        bf16x16 a = *reinterpret_cast<const bf16x16*>(Pw + (size_t)(m0 + lm) * C_ + k0);
        bf16x16 bfr[4];
        #pragma unroll
        for (int j = 0; j < 4; ++j)
            bfr[j] = *reinterpret_cast<const bf16x16*>(
                ab + (size_t)(n0 + j * 16 + lm) * C_ + k0);
        #pragma unroll
        for (int j = 0; j < 4; ++j)
            acc[j] = wmma_bf16(a, bfr[j], acc[j]);
    }

    #pragma unroll
    for (int j = 0; j < 4; ++j) {
        #pragma unroll
        for (int r = 0; r < 8; ++r) {
            const int m = m0 + r + 8 * hf;
            const int n = n0 + j * 16 + lm;
            const size_t idx = ((size_t)batch * C_ + m) * L_ + n;
            out[idx] = acc[j][r] + bias[m] + xres[idx];
        }
    }
}

// ---------------------------------------------------------------------------
// launch
// ---------------------------------------------------------------------------
extern "C" void kernel_launch(void* const* d_in, const int* in_sizes, int n_in,
                              void* d_out, int out_size, void* d_ws, size_t ws_size,
                              hipStream_t stream) {
    const float* x      = (const float*)d_in[0];
    const float* qkv_w  = (const float*)d_in[1];
    const float* proj_w = (const float*)d_in[2];
    const float* proj_b = (const float*)d_in[3];
    const float* gammap = (const float*)d_in[4];
    const float* betap  = (const float*)d_in[5];
    float* out = (float*)d_out;

    const size_t MB = 1u << 20;
    char* ws = (char*)d_ws;
    bf16* xnT   = (bf16*)(ws +  0 * MB);
    bf16* qT    = (bf16*)(ws +  8 * MB);
    bf16* kT    = (bf16*)(ws + 16 * MB);
    bf16* vv    = (bf16*)(ws + 24 * MB);
    bf16* attnT = (bf16*)(ws + 32 * MB);
    bf16* wqkv  = (bf16*)(ws + 40 * MB);
    bf16* wproj = (bf16*)(ws + 41 * MB);

    cvt_bf16_kernel<<<(3 * C_ * C_ + 255) / 256, 256, 0, stream>>>(qkv_w, wqkv, 3 * C_ * C_);
    cvt_bf16_kernel<<<(C_ * C_ + 255) / 256, 256, 0, stream>>>(proj_w, wproj, C_ * C_);

    gn_kernel<<<B_ * G_, 256, 0, stream>>>(x, gammap, betap, xnT);
    qkv_gemm_kernel<<<1536, 256, 0, stream>>>(wqkv, xnT, qT, kT, vv);
    attn_kernel<<<B_ * H_ * (L_ / 64), 128, 0, stream>>>(qT, kT, vv, attnT);
    proj_gemm_kernel<<<512, 256, 0, stream>>>(wproj, attnT, proj_b, x, out);
}
